// NAMCell_12189117186746
// MI455X (gfx1250) — matrix operand order repaired and checked by reading, not compiled, not run on hardware
//
#include <hip/hip_runtime.h>
#include <math.h>

// NAMCell forward for MI455X (gfx1250, wave32, WMMA).
// Big GEMMs + attention use v_wmma_f32_16x16x32_f16 (f16 in, f32 accum):
// f16 WMMA makes the GEMM side faster than the ~1GB of HBM/L2 traffic
// (memory tensor 134MB swept ~4x), i.e. bandwidth-bound -- the right
// operating point at 23.3 TB/s. Bandwidth-bound memory-tensor passes stay
// scalar fp32 with coalesced float4 access.
//
// Input order assumption (setup_inputs dict order, params in insertion order):
//  0 hidden 1 mask 2 memory 3 memory_usage 4 read_weights 5 write_weights
//  6 accumulated_result 7 accumulated_valid 8 steps 9.. params (w,b pairs in
//  dims order, then an_g, an_b, fn_g, fn_b).

typedef __attribute__((ext_vector_type(16))) _Float16 v16h;
typedef __attribute__((ext_vector_type(8)))  float    v8f;

#ifndef EPSF
#define EPSF 1e-6f
#endif

constexpr int ACT_NONE = 0;
constexpr int ACT_GELU = 1;
constexpr int ACT_RELU = 2;
constexpr int ACT_SIG  = 3;

// problem sizes
constexpr int Bb  = 64;
constexpr int Ll  = 128;
constexpr int Hh  = 512;
constexpr int Mm  = 1024;
constexpr int NHh = 8;
constexpr int DHh = 64;

__device__ __forceinline__ float sigmoidf_(float x){ return 1.f/(1.f+expf(-x)); }
__device__ __forceinline__ float geluf_(float x){
  return 0.5f*x*(1.f + tanhf(0.7978845608028654f*(x + 0.044715f*x*x*x)));
}
__device__ __forceinline__ float softplusf_(float x){ return (x > 20.f) ? x : log1pf(expf(x)); }
__device__ __forceinline__ float clip10_(float x){ return fminf(fmaxf(x, -1e10f), 1e10f); }

__device__ __forceinline__ float wave_sum(float v){
#pragma unroll
  for (int o=16;o>=1;o>>=1) v += __shfl_xor(v, o, 32);
  return v;
}
__device__ __forceinline__ float wave_max(float v){
#pragma unroll
  for (int o=16;o>=1;o>>=1) v = fmaxf(v, __shfl_xor(v, o, 32));
  return v;
}

// Build a 16x32 f16 A-fragment (wave32 WMMA; contiguous k order, not the ISA table order) from an f32 row.
// lane halves: K chunk [kbase+16h, +16) (contiguous).
__device__ __forceinline__ v16h a_frag_f32row(const float* rowptr, int kbase, int half){
  v16h a;
  const float* p0 = rowptr + kbase + 16*half;
  const float* p1 = p0 + 8;
#pragma unroll
  for (int e=0;e<8;++e){ a[e]=(_Float16)p0[e]; a[8+e]=(_Float16)p1[e]; }
  return a;
}

// ---------------------------------------------------------------------------
// Generic WMMA GEMM: C[M,N] = act(A[M,K] @ W[K,N] + bias) (+ Res)
// Block: 256 threads = 8 waves, tile 128x16 (wave w -> rows m0+16w..+15).
// B tile (32x16) staged in LDS in exact B-fragment order: lane n reads 16
// contiguous f16 (one 32B ds read) per k-step.
// ---------------------------------------------------------------------------
template<int ACT, bool HAS_RES>
__global__ __launch_bounds__(256) void gemm_wmma_kernel(
    const float* __restrict__ A, int lda,
    const float* __restrict__ W,          // K x N row-major
    const float* __restrict__ bias,       // N
    const float* __restrict__ Res, int ldr,
    float* __restrict__ C, int ldc,
    int M, int N, int K)
{
  __shared__ __align__(32) _Float16 bt[16*48];   // [n][k], row pad 48
  const int tid  = threadIdx.x;
  const int wave = tid >> 5;
  const int lane = tid & 31;
  const int half = (lane >> 4) & 1;
  const int ln   = lane & 15;
  const int m0   = blockIdx.x * 128;
  const int n0   = blockIdx.y * 16;
  const int arowi = m0 + wave*16 + ln;
  const bool rvalid = arowi < M;
  const float* arow = A + (size_t)(rvalid ? arowi : 0) * lda;

  v8f acc = {0.f,0.f,0.f,0.f,0.f,0.f,0.f,0.f};

  for (int k0 = 0; k0 < K; k0 += 32) {
    // cooperative stage of W[k0..k0+31][n0..n0+15] -> bt[n][k]
    {
      int kk = tid >> 4, nn = tid & 15;
      bt[nn*48 + kk]      = (_Float16)W[(size_t)(k0+kk)*N + n0 + nn];
      bt[nn*48 + kk + 16] = (_Float16)W[(size_t)(k0+kk+16)*N + n0 + nn];
    }
    __syncthreads();

    v16h a;
    if (rvalid) {
      a = a_frag_f32row(arow, k0, half);
      if (k0 + 32 < K) __builtin_prefetch(arow + k0 + 32, 0, 1);
    } else {
#pragma unroll
      for (int e=0;e<16;++e) a[e] = (_Float16)0.f;
    }
    v16h b = *(const v16h*)&bt[ln*48 + 16*half];
    acc = __builtin_amdgcn_wmma_f32_16x16x32_f16(false, a, false, b, (short)0, acc,
                                                 false, false);
    __syncthreads();
  }

  const int col = n0 + ln;
  const float bv = bias ? bias[col] : 0.f;
#pragma unroll
  for (int r = 0; r < 8; ++r) {
    int orow = m0 + wave*16 + r + 8*half;
    if (orow < M) {
      float v = acc[r] + bv;
      if (ACT == ACT_GELU) v = geluf_(v);
      else if (ACT == ACT_RELU) v = fmaxf(v, 0.f);
      else if (ACT == ACT_SIG)  v = sigmoidf_(v);
      if (HAS_RES) v += Res[(size_t)orow*ldr + col];
      C[(size_t)orow*ldc + col] = v;
    }
  }
}

// ---------------------------------------------------------------------------
// LayerNorm over H=512, one wave per row.
// ---------------------------------------------------------------------------
__global__ __launch_bounds__(256) void ln_kernel(const float* __restrict__ x,
                                                 const float* __restrict__ g,
                                                 const float* __restrict__ bb,
                                                 float* __restrict__ y, int nrows)
{
  int row = blockIdx.x*8 + (threadIdx.x >> 5);
  int lane = threadIdx.x & 31;
  if (row >= nrows) return;
  const float* xr = x + (size_t)row*Hh;
  float s = 0.f, s2 = 0.f;
  for (int i = lane; i < Hh; i += 32) { float v = xr[i]; s += v; s2 += v*v; }
  s = wave_sum(s); s2 = wave_sum(s2);
  float m   = s * (1.f/Hh);
  float var = s2 * (1.f/Hh) - m*m;
  float inv = rsqrtf(var + EPSF);
  float* yr = y + (size_t)row*Hh;
  for (int i = lane; i < Hh; i += 32) yr[i] = (xr[i]-m)*inv*g[i] + bb[i];
}

// ---------------------------------------------------------------------------
// link / cumulative-log kernel: one block per batch.
// ---------------------------------------------------------------------------
__global__ __launch_bounds__(128) void s_cl_kernel(const float* __restrict__ gq,
                                                   const float* __restrict__ gk,
                                                   const float* __restrict__ mask,
                                                   float* __restrict__ cl)
{
  __shared__ float llog[Ll];
  int b = blockIdx.x, t = threadIdx.x;
  if (t < Ll-1) {
    const float* a = gq + ((size_t)b*Ll + t)*Hh;
    const float* c = gk + ((size_t)b*Ll + t + 1)*Hh;
    float s = 0.f;
    for (int h = 0; h < Hh; ++h) s += a[h]*c[h];
    s *= 0.04419417382415922f;   // 1/sqrt(512)
    float link = sigmoidf_(s) * mask[b*Ll+t] * mask[b*Ll+t+1];
    llog[t] = logf(link + 1e-9f);
  }
  __syncthreads();
  if (t == 0) {
    float run = 0.f;
    cl[(size_t)b*Ll] = 0.f;
    for (int i = 1; i < Ll; ++i) { run += llog[i-1]; cl[(size_t)b*Ll + i] = run; }
  }
}

// ---------------------------------------------------------------------------
// Attention, one block per (b,h). Q from global as A-frags; K,V staged to LDS
// f16; S=QK^T via WMMA; softmax * group_prob renorm; attn=P@V via WMMA.
// ---------------------------------------------------------------------------
constexpr int KROW = 80;    // k_lds row pad (halves), 160B (32B aligned)
constexpr int VROW = 144;   // vt_lds row pad
constexpr int SROW = 132;   // s_lds row pad (f32)
constexpr size_t ATTN_SMEM = (size_t)Ll*KROW*2 + (size_t)DHh*VROW*2
                           + (size_t)Ll*SROW*4 + Ll*4 + Ll*4;  // 107,520 B

__global__ __launch_bounds__(256) void attn_kernel(
    const float* __restrict__ q, const float* __restrict__ k,
    const float* __restrict__ v, const float* __restrict__ mask,
    const float* __restrict__ cl, float* __restrict__ attn_out)
{
  extern __shared__ char smem[];
  _Float16* k_lds  = (_Float16*)smem;                              // [j][d]
  _Float16* vt_lds = (_Float16*)(smem + (size_t)Ll*KROW*2);        // [d][j]
  float*    s_lds  = (float*)(smem + (size_t)Ll*KROW*2 + (size_t)DHh*VROW*2);
  float*    cl_s   = (float*)((char*)s_lds + (size_t)Ll*SROW*4);
  float*    tm_s   = cl_s + Ll;

  const int b = blockIdx.x, h = blockIdx.y;
  const int tid = threadIdx.x, wave = tid >> 5, lane = tid & 31;
  const int half = (lane >> 4) & 1, ln = lane & 15;

  for (int idx = tid; idx < Ll*DHh; idx += 256) {
    int j = idx >> 6, d = idx & 63;
    size_t g = ((size_t)(b*Ll + j)*NHh + h)*DHh + d;
    k_lds[j*KROW + d]  = (_Float16)k[g];
    vt_lds[d*VROW + j] = (_Float16)v[g];
  }
  for (int idx = tid; idx < Ll; idx += 256) {
    cl_s[idx] = cl[(size_t)b*Ll + idx];
    tm_s[idx] = mask[(size_t)b*Ll + idx];
  }
  __syncthreads();

  // S = Q K^T * scale + mask-bias
  const float scale = 0.125f;  // 1/sqrt(64)
  const float* qrow = q + ((size_t)(b*Ll + wave*16 + ln)*NHh + h)*DHh;
  v16h aq0 = a_frag_f32row(qrow, 0, half);
  v16h aq1 = a_frag_f32row(qrow, 32, half);
#pragma unroll
  for (int nt = 0; nt < 8; ++nt) {
    v8f acc = {0.f,0.f,0.f,0.f,0.f,0.f,0.f,0.f};
    v16h bf0 = *(const v16h*)&k_lds[(nt*16+ln)*KROW + 16*half];
    acc = __builtin_amdgcn_wmma_f32_16x16x32_f16(false, aq0, false, bf0, (short)0, acc, false, false);
    v16h bf1 = *(const v16h*)&k_lds[(nt*16+ln)*KROW + 16*half + 32];
    acc = __builtin_amdgcn_wmma_f32_16x16x32_f16(false, aq1, false, bf1, (short)0, acc, false, false);
#pragma unroll
    for (int r = 0; r < 8; ++r) {
      int i = wave*16 + r + 8*half;
      int j = nt*16 + ln;
      s_lds[i*SROW + j] = acc[r]*scale + (1.f - tm_s[j])*(-1e9f);
    }
  }
  __syncthreads();

  // softmax * group_prob, renormalized: p = e*g / (sum(e*g) + EPS*sum(e))
  for (int rr = 0; rr < 16; ++rr) {
    int i = wave*16 + rr;
    float cli = cl_s[i];
    float sv[4], gv[4];
    float mx = -1e30f;
#pragma unroll
    for (int c = 0; c < 4; ++c) { sv[c] = s_lds[i*SROW + lane + 32*c]; mx = fmaxf(mx, sv[c]); }
    mx = wave_max(mx);
    float se = 0.f, seg = 0.f;
#pragma unroll
    for (int c = 0; c < 4; ++c) {
      int j = lane + 32*c;
      float e = expf(sv[c] - mx);
      float g = expf(-fabsf(cli - cl_s[j]));
      gv[c] = e*g; se += e; seg += gv[c];
    }
    se = wave_sum(se); seg = wave_sum(seg);
    float inv = 1.f / (seg + EPSF*se);
#pragma unroll
    for (int c = 0; c < 4; ++c) s_lds[i*SROW + lane + 32*c] = gv[c]*inv;
  }
  __syncthreads();

  // attn = P @ V
  v8f accd[4];
#pragma unroll
  for (int nt = 0; nt < 4; ++nt) accd[nt] = (v8f){0.f,0.f,0.f,0.f,0.f,0.f,0.f,0.f};
  const float* prow = &s_lds[(wave*16 + ln)*SROW];
#pragma unroll
  for (int ks = 0; ks < 4; ++ks) {
    v16h ap = a_frag_f32row(prow, ks*32, half);
#pragma unroll
    for (int nt = 0; nt < 4; ++nt) {
      v16h bf = *(const v16h*)&vt_lds[(nt*16+ln)*VROW + 16*half + 32*ks];
      accd[nt] = __builtin_amdgcn_wmma_f32_16x16x32_f16(false, ap, false, bf, (short)0,
                                                        accd[nt], false, false);
    }
  }
#pragma unroll
  for (int nt = 0; nt < 4; ++nt)
#pragma unroll
    for (int r = 0; r < 8; ++r) {
      int i = wave*16 + r + 8*half;
      int d = nt*16 + ln;
      attn_out[(size_t)(b*Ll + i)*Hh + h*DHh + d] = accd[nt][r];
    }
}

// ---------------------------------------------------------------------------
// rs = hidden@red_w + red_b + mask-bias; softmax over L; pooled = sum rw*hidden
// ---------------------------------------------------------------------------
__global__ __launch_bounds__(256) void rs_pool_kernel(const float* __restrict__ hidden,
                                                      const float* __restrict__ redw,
                                                      const float* __restrict__ redb,
                                                      const float* __restrict__ mask,
                                                      float* __restrict__ pooled, int ldp)
{
  __shared__ float rs[Ll];
  __shared__ float rw[Ll];
  int b = blockIdx.x, tid = threadIdx.x, wave = tid >> 5, lane = tid & 31;
  for (int l = wave; l < Ll; l += 8) {
    const float* hr = hidden + ((size_t)b*Ll + l)*Hh;
    float s = 0.f;
    for (int i = lane; i < Hh; i += 32) s += hr[i]*redw[i];
    s = wave_sum(s);
    if (lane == 0) rs[l] = s + redb[0] + (1.f - mask[b*Ll + l])*(-1e9f);
  }
  __syncthreads();
  if (tid == 0) {
    float mx = -1e30f;
    for (int l = 0; l < Ll; ++l) mx = fmaxf(mx, rs[l]);
    float sum = 0.f;
    for (int l = 0; l < Ll; ++l) { rw[l] = expf(rs[l]-mx); sum += rw[l]; }
    float inv = 1.f/sum;
    for (int l = 0; l < Ll; ++l) rw[l] *= inv;
  }
  __syncthreads();
  for (int hh = tid; hh < Hh; hh += 256) {
    float acc = 0.f;
    for (int l = 0; l < Ll; ++l) acc += rw[l]*hidden[((size_t)b*Ll + l)*Hh + hh];
    pooled[(size_t)b*ldp + hh] = acc;
  }
}

// ---------------------------------------------------------------------------
// per-row L2 norms of the memory tensor (computed once, reused 3x)
// ---------------------------------------------------------------------------
__global__ __launch_bounds__(256) void memnorm_kernel(const float* __restrict__ mem,
                                                      float* __restrict__ norms, int nrows)
{
  int row = blockIdx.x*8 + (threadIdx.x >> 5);
  int lane = threadIdx.x & 31;
  if (row >= nrows) return;
  const float* r = mem + (size_t)row*Hh;
  float s = 0.f;
  for (int i = lane; i < Hh; i += 32) { float v = r[i]; s += v*v; }
  s = wave_sum(s);
  if (lane == 0) norms[row] = sqrtf(s);
}

// ---------------------------------------------------------------------------
// content addressing: w = softmax(softplus(betavec.bw + bbias) * cos-sim),
// optional weighted read written to up to two destinations. One block per b.
// ---------------------------------------------------------------------------
__global__ __launch_bounds__(256) void content_kernel(
    const float* __restrict__ keybase, int ldk,
    const float* __restrict__ betavec, int ldbv,
    const float* __restrict__ bw, const float* __restrict__ bbias,
    const float* __restrict__ mem, const float* __restrict__ norms,
    float* __restrict__ wout,
    float* __restrict__ rd1, int ldr1,
    float* __restrict__ rd2, int ldr2)
{
  __shared__ float keys[Hh];
  __shared__ float part[256];
  __shared__ float part2[256];
  __shared__ float sims[Mm];
  __shared__ float stats[2];
  int b = blockIdx.x, tid = threadIdx.x, wave = tid >> 5, lane = tid & 31;
  const float* key  = keybase + (size_t)b*ldk;
  const float* bvec = betavec + (size_t)b*ldbv;

  float p1 = 0.f, p2 = 0.f;
  for (int i = tid; i < Hh; i += 256) {
    float kv = key[i]; keys[i] = kv;
    p1 += kv*kv;
    p2 += bvec[i]*bw[i];
  }
  part[tid] = p1; part2[tid] = p2;
  __syncthreads();
  if (tid == 0) {
    float s1 = 0.f, s2 = 0.f;
    for (int i = 0; i < 256; ++i) { s1 += part[i]; s2 += part2[i]; }
    stats[0] = 1.f/(sqrtf(s1) + EPSF);
    stats[1] = softplusf_(s2 + bbias[0]);
  }
  __syncthreads();
  float kninv = stats[0], beta = stats[1];
  const float* memb = mem + (size_t)b*Mm*Hh;

  for (int m = wave; m < Mm; m += 8) {
    const float* mr = memb + (size_t)m*Hh;
    float s = 0.f;
    for (int i = lane; i < Hh; i += 32) s += mr[i]*keys[i];
    s = wave_sum(s);
    if (lane == 0) sims[m] = beta * (s * kninv / (norms[(size_t)b*Mm + m] + EPSF));
  }
  __syncthreads();

  float lm = -1e30f;
  for (int m = tid; m < Mm; m += 256) lm = fmaxf(lm, sims[m]);
  part[tid] = lm;
  __syncthreads();
  if (tid == 0) { float mx = -1e30f; for (int i = 0; i < 256; ++i) mx = fmaxf(mx, part[i]); stats[0] = mx; }
  __syncthreads();
  float mx = stats[0], ls = 0.f;
  for (int m = tid; m < Mm; m += 256) { float e = expf(sims[m]-mx); sims[m] = e; ls += e; }
  part[tid] = ls;
  __syncthreads();
  if (tid == 0) { float s = 0.f; for (int i = 0; i < 256; ++i) s += part[i]; stats[1] = 1.f/s; }
  __syncthreads();
  float inv = stats[1];
  for (int m = tid; m < Mm; m += 256) {
    float w = sims[m]*inv; sims[m] = w;
    wout[(size_t)b*Mm + m] = w;
  }
  __syncthreads();

  if (rd1) {
    for (int hh = tid; hh < Hh; hh += 256) {
      float acc = 0.f;
      for (int m = 0; m < Mm; ++m) acc += sims[m]*memb[(size_t)m*Hh + hh];
      rd1[(size_t)b*ldr1 + hh] = acc;
      if (rd2) rd2[(size_t)b*ldr2 + hh] = acc;
    }
  }
}

// ---------------------------------------------------------------------------
// new_memory = mem*(1 - ww*erase) + ww*add   (float4, bandwidth-bound)
// ---------------------------------------------------------------------------
__global__ __launch_bounds__(256) void memupdate_kernel(const float* __restrict__ mem,
                                                        const float* __restrict__ ww,
                                                        const float* __restrict__ erase,
                                                        const float* __restrict__ addv,
                                                        float* __restrict__ out)
{
  size_t base = ((size_t)blockIdx.x*256 + threadIdx.x) * 4;
  constexpr size_t total = (size_t)Bb*Mm*Hh;
  if (base >= total) return;
  size_t bm = base / Hh;
  int hh = (int)(base & (Hh-1));
  int b = (int)(bm >> 10);
  float w = ww[bm];
  float4 e  = *(const float4*)&erase[(size_t)b*Hh + hh];
  float4 ad = *(const float4*)&addv [(size_t)b*Hh + hh];
  float4 mm = *(const float4*)&mem[base];
  float4 r;
  r.x = mm.x*(1.f - w*e.x) + w*ad.x;
  r.y = mm.y*(1.f - w*e.y) + w*ad.y;
  r.z = mm.z*(1.f - w*e.z) + w*ad.z;
  r.w = mm.w*(1.f - w*e.w) + w*ad.w;
  *(float4*)&out[base] = r;
}

__global__ __launch_bounds__(256) void usage_kernel(const float* __restrict__ u,
                                                    const float* __restrict__ ww,
                                                    float* __restrict__ out)
{
  int i = blockIdx.x*256 + threadIdx.x;
  if (i < Bb*Mm) { float a = u[i], w = ww[i]; out[i] = a + w - a*w; }
}

// ---------------------------------------------------------------------------
// scalar arithmetic head: a,b,op softmax, result/valid, renc/venc rows.
// ---------------------------------------------------------------------------
__global__ __launch_bounds__(256) void scalar_ops_kernel(
    const float* __restrict__ leftf, const float* __restrict__ rightf,
    const float* __restrict__ pooled, int ldp,
    const float* __restrict__ lnum_w, const float* __restrict__ lnum_b,
    const float* __restrict__ rnum_w, const float* __restrict__ rnum_b,
    const float* __restrict__ op_w, const float* __restrict__ op_b,
    const float* __restrict__ renc_w, const float* __restrict__ renc_b,
    const float* __restrict__ venc_w, const float* __restrict__ venc_b,
    const float* __restrict__ acc_res,
    float* __restrict__ out_result, float* __restrict__ out_valid,
    float* __restrict__ out_oplog, float* __restrict__ supd_in)
{
  __shared__ float pa[256], pb[256];
  __shared__ float po[4][256];
  __shared__ float sres[2];
  int b = blockIdx.x, tid = threadIdx.x;
  const float* lf = leftf  + (size_t)b*Hh;
  const float* rf = rightf + (size_t)b*Hh;
  const float* pl = pooled + (size_t)b*ldp;
  float a1 = 0.f, b1 = 0.f, o0 = 0.f, o1 = 0.f, o2 = 0.f, o3 = 0.f;
  for (int i = tid; i < Hh; i += 256) {
    a1 += lf[i]*lnum_w[i];
    b1 += rf[i]*rnum_w[i];
    float p = pl[i];
    o0 += p*op_w[i*4+0]; o1 += p*op_w[i*4+1]; o2 += p*op_w[i*4+2]; o3 += p*op_w[i*4+3];
  }
  pa[tid] = a1; pb[tid] = b1;
  po[0][tid] = o0; po[1][tid] = o1; po[2][tid] = o2; po[3][tid] = o3;
  __syncthreads();
  if (tid == 0) {
    float A = 0.f, Bv = 0.f, O[4] = {0.f,0.f,0.f,0.f};
    for (int i = 0; i < 256; ++i) {
      A += pa[i]; Bv += pb[i];
      O[0] += po[0][i]; O[1] += po[1][i]; O[2] += po[2][i]; O[3] += po[3][i];
    }
    A  = clip10_(A  + lnum_b[0]);
    Bv = clip10_(Bv + rnum_b[0]);
    float lg[4], mx = -1e30f;
    for (int j = 0; j < 4; ++j) { lg[j] = O[j] + op_b[j]; out_oplog[b*4+j] = lg[j]; mx = fmaxf(mx, lg[j]); }
    float pr[4], sum = 0.f;
    for (int j = 0; j < 4; ++j) { pr[j] = expf(lg[j]-mx); sum += pr[j]; }
    for (int j = 0; j < 4; ++j) pr[j] /= sum;
    float absb = fabsf(Bv);
    float dv = (absb > 1e-7f) ? 1.f : 0.f;
    float divv = A / ((absb > 1e-7f) ? Bv : 1.f) * dv;
    float res[4] = {A+Bv, A-Bv, A*Bv, divv};
    float val[4] = {1.f, 1.f, 1.f, dv};
    float result = 0.f, valid = 0.f;
    for (int j = 0; j < 4; ++j) { result += pr[j]*res[j]; valid += pr[j]*val[j]; }
    out_result[b] = result; out_valid[b] = valid;
    sres[0] = result; sres[1] = valid;
  }
  __syncthreads();
  float result = sres[0], valid = sres[1], ar = acc_res[b];
  float* srow = supd_in + (size_t)b*2560;
  for (int hh = tid; hh < Hh; hh += 256) {
    srow[512  + hh] = result*renc_w[hh] + ar*renc_w[Hh+hh] + renc_b[hh];
    srow[1024 + hh] = valid*venc_w[hh] + venc_b[hh];
  }
}

__global__ __launch_bounds__(256) void halt_kernel(const float* __restrict__ state,
                                                   const float* __restrict__ hw,
                                                   const float* __restrict__ hb,
                                                   float* __restrict__ out)
{
  __shared__ float p0[256], p1[256];
  int b = blockIdx.x, tid = threadIdx.x;
  const float* s = state + (size_t)b*Hh;
  float a = 0.f, c = 0.f;
  for (int i = tid; i < Hh; i += 256) { a += s[i]*hw[i*2]; c += s[i]*hw[i*2+1]; }
  p0[tid] = a; p1[tid] = c;
  __syncthreads();
  if (tid == 0) {
    float sa = 0.f, sc = 0.f;
    for (int i = 0; i < 256; ++i) { sa += p0[i]; sc += p1[i]; }
    out[b*2+0] = sa + hb[0];
    out[b*2+1] = sc + hb[1];
  }
}

// ---------------------------------------------------------------------------
// host launch
// ---------------------------------------------------------------------------
enum ParamIdx {
  GQ_W=0, GQ_B, GK_W, GK_B, Q_W, Q_B, K_W, K_B, V_W, V_B, O_W, O_B,
  FFN1_W, FFN1_B, FFN2_W, FFN2_B, RED_W, RED_B, LEFT_W, LEFT_B, RIGHT_W, RIGHT_B,
  LNUM_W, LNUM_B, RNUM_W, RNUM_B, OP_W, OP_B, R0K_W, R0K_B, R0B_W, R0B_B,
  R1K_W, R1K_B, R1B_W, R1B_B, CTRL_W, CTRL_B, RENC_W, RENC_B, VENC_W, VENC_B,
  SUPD_W, SUPD_B, WK_W, WK_B, WB_W, WB_B, WE_W, WE_B, WA_W, WA_B, HALT_W, HALT_B,
  AN_G, AN_B, FN_G, FN_B
};

extern "C" void kernel_launch(void* const* d_in, const int* in_sizes, int n_in,
                              void* d_out, int out_size, void* d_ws, size_t ws_size,
                              hipStream_t stream)
{
  (void)in_sizes; (void)n_in; (void)out_size; (void)ws_size;

  const float* hidden  = (const float*)d_in[0];
  const float* mask    = (const float*)d_in[1];
  const float* memory  = (const float*)d_in[2];
  const float* musage  = (const float*)d_in[3];
  const float* acc_res = (const float*)d_in[6];
  const float* const* P = (const float* const*)(d_in + 9);

  // ---- workspace layout (floats) ----
  const size_t SZ = (size_t)Bb*Ll*Hh;            // 4,194,304
  float* ws   = (float*)d_ws;
  float* bufA = ws;                              // gq   -> attn
  float* bufB = ws + SZ;                         // gk   -> hn2
  float* bufC = ws + 2*SZ;                       // hn   -> hidden2
  float* bufD = ws + 3*SZ;                       // q,k,v -> ffn intermediate (4*SZ)
  float* q_buf = bufD;
  float* k_buf = bufD + SZ;
  float* v_buf = bufD + 2*SZ;
  float* sm    = ws + 7*SZ;
  float* cl_buf  = sm;                 sm += (size_t)Bb*Ll;
  float* ctrl_in = sm;                 sm += (size_t)Bb*1536;
  float* supd_in = sm;                 sm += (size_t)Bb*2560;
  float* norms   = sm;                 sm += (size_t)Bb*Mm;
  float* leftf   = sm;                 sm += (size_t)Bb*Hh;
  float* rightf  = sm;                 sm += (size_t)Bb*Hh;
  float* key0    = sm;                 sm += (size_t)Bb*Hh;
  float* key1    = sm;                 sm += (size_t)Bb*Hh;
  float* state   = sm;                 sm += (size_t)Bb*Hh;
  float* wkey    = sm;                 sm += (size_t)Bb*Hh;
  float* eras    = sm;                 sm += (size_t)Bb*Hh;
  float* addv    = sm;                 sm += (size_t)Bb*Hh;

  // ---- output layout (floats, concatenated in return order) ----
  float* out = (float*)d_out;
  float* o_newmem = out;                                   // B*M*H
  float* o_result = o_newmem + (size_t)Bb*Mm*Hh;           // B
  float* o_valid  = o_result + Bb;
  float* o_oplog  = o_valid  + Bb;                         // B*4
  float* o_halt   = o_oplog  + Bb*4;                       // B*2
  float* o_hidden = o_halt   + Bb*2;                       // B*L*H
  float* o_usage  = o_hidden + SZ;                         // B*M
  float* o_rw     = o_usage  + (size_t)Bb*Mm;              // R*B*M
  float* o_ww     = o_rw     + (size_t)2*Bb*Mm;            // B*M

  const dim3 blk(256);
  auto grd = [](int M, int N){ return dim3((unsigned)((M+127)/128), (unsigned)(N/16)); };
  const int ML = Bb*Ll;   // 8192

  // 1-2: grouping projections
  gemm_wmma_kernel<ACT_NONE,false><<<grd(ML,Hh), blk, 0, stream>>>(
      hidden, Hh, P[GQ_W], P[GQ_B], nullptr, 0, bufA, Hh, ML, Hh, Hh);
  gemm_wmma_kernel<ACT_NONE,false><<<grd(ML,Hh), blk, 0, stream>>>(
      hidden, Hh, P[GK_W], P[GK_B], nullptr, 0, bufB, Hh, ML, Hh, Hh);
  // 3: link -> cumulative log
  s_cl_kernel<<<Bb, 128, 0, stream>>>(bufA, bufB, mask, cl_buf);
  // 4: LN -> hn
  ln_kernel<<<ML/8, blk, 0, stream>>>(hidden, P[AN_G], P[AN_B], bufC, ML);
  // 5: q,k,v
  gemm_wmma_kernel<ACT_NONE,false><<<grd(ML,Hh), blk, 0, stream>>>(
      bufC, Hh, P[Q_W], P[Q_B], nullptr, 0, q_buf, Hh, ML, Hh, Hh);
  gemm_wmma_kernel<ACT_NONE,false><<<grd(ML,Hh), blk, 0, stream>>>(
      bufC, Hh, P[K_W], P[K_B], nullptr, 0, k_buf, Hh, ML, Hh, Hh);
  gemm_wmma_kernel<ACT_NONE,false><<<grd(ML,Hh), blk, 0, stream>>>(
      bufC, Hh, P[V_W], P[V_B], nullptr, 0, v_buf, Hh, ML, Hh, Hh);
  // 6: attention (bufA reused)
  attn_kernel<<<dim3(Bb, NHh), blk, ATTN_SMEM, stream>>>(q_buf, k_buf, v_buf, mask, cl_buf, bufA);
  // 7: o-proj + residual -> hidden2 (bufC)
  gemm_wmma_kernel<ACT_NONE,true><<<grd(ML,Hh), blk, 0, stream>>>(
      bufA, Hh, P[O_W], P[O_B], hidden, Hh, bufC, Hh, ML, Hh, Hh);
  // 8: LN -> hn2 (bufB)
  ln_kernel<<<ML/8, blk, 0, stream>>>(bufC, P[FN_G], P[FN_B], bufB, ML);
  // 9-10: FFN (gelu) + residual -> final hidden (written straight into d_out)
  gemm_wmma_kernel<ACT_GELU,false><<<grd(ML,2048), blk, 0, stream>>>(
      bufB, Hh, P[FFN1_W], P[FFN1_B], nullptr, 0, bufD, 2048, ML, 2048, Hh);
  gemm_wmma_kernel<ACT_NONE,true><<<grd(ML,Hh), blk, 0, stream>>>(
      bufD, 2048, P[FFN2_W], P[FFN2_B], bufC, Hh, o_hidden, Hh, ML, Hh, 2048);
  // 11: rs softmax + pooled -> ctrl_in[:,0:512]
  rs_pool_kernel<<<Bb, blk, 0, stream>>>(o_hidden, P[RED_W], P[RED_B], mask, ctrl_in, 1536);
  // 12-13: small M=64 GEMMs off pooled
  gemm_wmma_kernel<ACT_NONE,false><<<grd(Bb,Hh), blk, 0, stream>>>(
      ctrl_in, 1536, P[LEFT_W],  P[LEFT_B],  nullptr, 0, leftf,  Hh, Bb, Hh, Hh);
  gemm_wmma_kernel<ACT_NONE,false><<<grd(Bb,Hh), blk, 0, stream>>>(
      ctrl_in, 1536, P[RIGHT_W], P[RIGHT_B], nullptr, 0, rightf, Hh, Bb, Hh, Hh);
  gemm_wmma_kernel<ACT_NONE,false><<<grd(Bb,Hh), blk, 0, stream>>>(
      ctrl_in, 1536, P[R0K_W], P[R0K_B], nullptr, 0, key0, Hh, Bb, Hh, Hh);
  gemm_wmma_kernel<ACT_NONE,false><<<grd(Bb,Hh), blk, 0, stream>>>(
      ctrl_in, 1536, P[R1K_W], P[R1K_B], nullptr, 0, key1, Hh, Bb, Hh, Hh);
  // 14: memory row norms (reused by all three addressings)
  memnorm_kernel<<<(Bb*Mm)/8, blk, 0, stream>>>(memory, norms, Bb*Mm);
  // 15-16: read addressing + reads (into both concat buffers)
  content_kernel<<<Bb, blk, 0, stream>>>(key0, Hh, ctrl_in, 1536, P[R0B_W], P[R0B_B],
      memory, norms, o_rw, ctrl_in + 512, 1536, supd_in + 1536, 2560);
  content_kernel<<<Bb, blk, 0, stream>>>(key1, Hh, ctrl_in, 1536, P[R1B_W], P[R1B_B],
      memory, norms, o_rw + (size_t)Bb*Mm, ctrl_in + 1024, 1536, supd_in + 2048, 2560);
  // 17: ctrl (relu) -> supd_in[:,0:512]
  gemm_wmma_kernel<ACT_RELU,false><<<grd(Bb,Hh), blk, 0, stream>>>(
      ctrl_in, 1536, P[CTRL_W], P[CTRL_B], nullptr, 0, supd_in, 2560, Bb, Hh, 1536);
  // 18: scalar arithmetic head + renc/venc -> supd_in[:,512:1536]
  scalar_ops_kernel<<<Bb, blk, 0, stream>>>(leftf, rightf, ctrl_in, 1536,
      P[LNUM_W], P[LNUM_B], P[RNUM_W], P[RNUM_B], P[OP_W], P[OP_B],
      P[RENC_W], P[RENC_B], P[VENC_W], P[VENC_B], acc_res,
      o_result, o_valid, o_oplog, supd_in);
  // 19: state
  gemm_wmma_kernel<ACT_NONE,false><<<grd(Bb,Hh), blk, 0, stream>>>(
      supd_in, 2560, P[SUPD_W], P[SUPD_B], nullptr, 0, state, Hh, Bb, Hh, 2560);
  // 20: write-head key / erase (sigmoid) / add
  gemm_wmma_kernel<ACT_NONE,false><<<grd(Bb,Hh), blk, 0, stream>>>(
      state, Hh, P[WK_W], P[WK_B], nullptr, 0, wkey, Hh, Bb, Hh, Hh);
  gemm_wmma_kernel<ACT_SIG,false><<<grd(Bb,Hh), blk, 0, stream>>>(
      state, Hh, P[WE_W], P[WE_B], nullptr, 0, eras, Hh, Bb, Hh, Hh);
  gemm_wmma_kernel<ACT_NONE,false><<<grd(Bb,Hh), blk, 0, stream>>>(
      state, Hh, P[WA_W], P[WA_B], nullptr, 0, addv, Hh, Bb, Hh, Hh);
  // 21: write addressing (no read)
  content_kernel<<<Bb, blk, 0, stream>>>(wkey, Hh, state, Hh, P[WB_W], P[WB_B],
      memory, norms, o_ww, nullptr, 0, nullptr, 0);
  // 22: memory update
  memupdate_kernel<<<(unsigned)(((size_t)Bb*Mm*Hh/4 + 255)/256), blk, 0, stream>>>(
      memory, o_ww, eras, addv, o_newmem);
  // 23: usage
  usage_kernel<<<(Bb*Mm + 255)/256, blk, 0, stream>>>(musage, o_ww, o_usage);
  // 24: halt
  halt_kernel<<<Bb, blk, 0, stream>>>(state, P[HALT_W], P[HALT_B], o_halt);
}
